// SparseMoE_40647570489877
// MI455X (gfx1250) — compile-verified, matrix operand order checked by reading, not run on hardware
//
#include <hip/hip_runtime.h>
#include <hip/hip_bf16.h>

// ---------------- problem constants ----------------
constexpr int E   = 8;     // experts
constexpr int TOPK= 2;
constexpr int D   = 768;
constexpr int H   = 2048;
constexpr int NT  = 8192;  // B*T tokens
constexpr int BM  = 128, BN = 128, BK = 32;
constexpr int TILES_M = NT / BM;          // 64
constexpr int ST  = 40;                   // LDS row stride (elems), breaks bank conflicts

typedef __attribute__((ext_vector_type(16))) __bf16 v16bf;
typedef __attribute__((ext_vector_type(8)))  float  v8f;
typedef int v4i __attribute__((vector_size(16)));   // matches builtin param type

// ---------------- CDNA5 async global->LDS path (guarded) ----------------
#if defined(__AMDGCN__) && __has_builtin(__builtin_amdgcn_global_load_async_to_lds_b128)
#define HAVE_ASYNC 1
#else
#define HAVE_ASYNC 0
#endif

#define AS_GLOBAL __attribute__((address_space(1)))
#define AS_LDS    __attribute__((address_space(3)))

static __device__ __forceinline__ void cp16_g2l(void* lds, const void* g) {
#if HAVE_ASYNC
  // emits global_load_async_to_lds_b128 (ASYNCcnt-tracked, bypasses VGPRs)
  __builtin_amdgcn_global_load_async_to_lds_b128((AS_GLOBAL v4i*)g,
                                                 (AS_LDS v4i*)lds, 0, 0);
#else
  *(uint4*)lds = *(const uint4*)g;
#endif
}

static __device__ __forceinline__ void wait_async0() {
#if HAVE_ASYNC
#if __has_builtin(__builtin_amdgcn_s_wait_asynccnt)
  __builtin_amdgcn_s_wait_asynccnt(0);
#else
  asm volatile("s_wait_asynccnt 0x0" ::: "memory");
#endif
#endif
}

static __device__ __forceinline__ v8f wmma_bf16(v16bf a, v16bf b, v8f c) {
  return __builtin_amdgcn_wmma_f32_16x16x32_bf16(false, a, false, b, (short)0, c,
                                                 false, false);
}

// Load one 16x32 bf16 fragment row-slice: lane holds 16 bf16 = two 16B chunks
// at element offsets kb and kb+16 of a 32-elem LDS row (kb = 0 or 8).
static __device__ __forceinline__ v16bf ld_frag(const __bf16* p, int kb) {
  union { uint4 u[2]; v16bf v; } r;
  r.u[0] = *(const uint4*)(p + kb);
  r.u[1] = *(const uint4*)(p + kb + 16);
  return r.v;
}

static __device__ __forceinline__ int imin(int a, int b) { return a < b ? a : b; }

// ---------------- conversion kernels ----------------
__global__ __launch_bounds__(256) void cvt_x_kernel(const float* __restrict__ x,
                                                    __bf16* __restrict__ xb) {
  size_t i = ((size_t)blockIdx.x * blockDim.x + threadIdx.x) * 8;
  float4 a = *(const float4*)(x + i);
  float4 b = *(const float4*)(x + i + 4);
  union { __bf16 h[8]; uint4 u; } o;
  o.h[0] = (__bf16)a.x; o.h[1] = (__bf16)a.y; o.h[2] = (__bf16)a.z; o.h[3] = (__bf16)a.w;
  o.h[4] = (__bf16)b.x; o.h[5] = (__bf16)b.y; o.h[6] = (__bf16)b.z; o.h[7] = (__bf16)b.w;
  *(uint4*)(xb + i) = o.u;
}

// in: fp32 [E][R][C]  ->  out: bf16 [E][C][R]  (transpose + convert)
__global__ __launch_bounds__(256) void transpose_cvt_kernel(const float* __restrict__ in,
                                                            __bf16* __restrict__ out,
                                                            int R, int C) {
  __shared__ float tile[32][33];
  const int eidx = blockIdx.z;
  const float* pin = in + (size_t)eidx * R * C;
  __bf16* pout = out + (size_t)eidx * R * C;
  int c0 = blockIdx.x * 32, r0 = blockIdx.y * 32;
  int tx = threadIdx.x, ty = threadIdx.y;   // 32 x 8
#pragma unroll
  for (int i = 0; i < 32; i += 8)
    tile[ty + i][tx] = pin[(size_t)(r0 + ty + i) * C + c0 + tx];
  __syncthreads();
#pragma unroll
  for (int i = 0; i < 32; i += 8)
    pout[(size_t)(c0 + ty + i) * R + r0 + tx] = (__bf16)tile[tx][ty + i];
}

// ---------------- router: one wave (32 lanes) per token ----------------
__global__ __launch_bounds__(256) void router_kernel(
    const float* __restrict__ x, const float* __restrict__ noise,
    const float* __restrict__ wg, const float* __restrict__ bg,
    const float* __restrict__ wn, const float* __restrict__ bn,
    int* __restrict__ counts, int* __restrict__ btok, float* __restrict__ bgate) {
  const int wave = threadIdx.x >> 5, lane = threadIdx.x & 31;
  const int t = blockIdx.x * 8 + wave;
  float ag[E] = {}, an[E] = {};
  const float* xr = x + (size_t)t * D;
  for (int d = lane; d < D; d += 32) {
    float xv = xr[d];
#pragma unroll
    for (int e = 0; e < E; ++e) {
      ag[e] = fmaf(xv, wg[d * E + e], ag[e]);
      an[e] = fmaf(xv, wn[d * E + e], an[e]);
    }
  }
#pragma unroll
  for (int off = 16; off > 0; off >>= 1)
#pragma unroll
    for (int e = 0; e < E; ++e) {
      ag[e] += __shfl_xor(ag[e], off);
      an[e] += __shfl_xor(an[e], off);
    }
  if (lane == 0) {
    float nv[E];
#pragma unroll
    for (int e = 0; e < E; ++e) {
      float lg = ag[e] + bg[e];
      float nl = an[e] + bn[e];
      float sp = nl > 20.f ? nl : __logf(1.f + __expf(nl));  // softplus
      nv[e] = lg + noise[(size_t)t * E + e] * sp;
    }
    int i0 = 0;
#pragma unroll
    for (int e = 1; e < E; ++e) if (nv[e] > nv[i0]) i0 = e;
    int i1 = (i0 == 0) ? 1 : 0;
#pragma unroll
    for (int e = 0; e < E; ++e) if (e != i0 && nv[e] > nv[i1]) i1 = e;
    float e1 = __expf(nv[i1] - nv[i0]);
    float inv = 1.f / (1.f + e1);
    float g0 = inv, g1 = e1 * inv;
    int p0 = atomicAdd(&counts[i0], 1);
    btok[i0 * NT + p0] = t; bgate[i0 * NT + p0] = g0;
    int p1 = atomicAdd(&counts[i1], 1);
    btok[i1 * NT + p1] = t; bgate[i1 * NT + p1] = g1;
  }
}

__global__ void offsets_kernel(const int* __restrict__ counts, int* __restrict__ offs) {
  if (threadIdx.x == 0) {
    int s = 0;
#pragma unroll
    for (int e = 0; e < E; ++e) { offs[e] = s; s += counts[e]; }
  }
}

// ---------------- GEMM1: h = silu(x@w1) * (x@w3), gathered rows ----------------
__global__ __launch_bounds__(256) void moe_gemm1(
    const __bf16* __restrict__ xb, const __bf16* __restrict__ w1b,
    const __bf16* __restrict__ w3b, const int* __restrict__ btok,
    const int* __restrict__ counts, const int* __restrict__ offs,
    __bf16* __restrict__ hb) {
  __shared__ __attribute__((aligned(16))) __bf16 sA [2][BM * ST];
  __shared__ __attribute__((aligned(16))) __bf16 sB1[2][BM * ST];
  __shared__ __attribute__((aligned(16))) __bf16 sB3[2][BM * ST];
  __shared__ int s_tok[BM];

  const int e    = blockIdx.x / TILES_M;
  const int row0 = (blockIdx.x % TILES_M) * BM;
  const int cnt  = counts[e];
  if (row0 >= cnt) return;
  const int base = offs[e];
  const int n0   = blockIdx.y * BN;
  const int tid  = threadIdx.x;

  for (int i = tid; i < BM; i += 256)
    s_tok[i] = (row0 + i < cnt) ? btok[e * NT + row0 + i] : 0;
  __syncthreads();

  const int wave = tid >> 5, lane = tid & 31;
  const int wm = wave & 3, wn = wave >> 2;      // 4 x 2 wave grid over 128x128
  const int lm = lane & 15, hi = lane >> 4;
  const int kb = hi * 8;
  v8f acc1[2][4] = {}; v8f acc3[2][4] = {};
  const __bf16* pw1 = w1b + (size_t)e * H * D;  // [H][D]
  const __bf16* pw3 = w3b + (size_t)e * H * D;

  // this thread's two fixed 16B chunks per tile: precompute all base pointers
  const int c0 = tid, c1 = tid + 256;
  const int r0c = c0 >> 2, k0c = (c0 & 3) * 8;
  const int r1c = c1 >> 2, k1c = (c1 & 3) * 8;
  const int dOff0 = r0c * ST + k0c, dOff1 = r1c * ST + k1c;
  const __bf16* gA0 = xb + (size_t)s_tok[r0c] * D + k0c;
  const __bf16* gA1 = xb + (size_t)s_tok[r1c] * D + k1c;
  const __bf16* gB10 = pw1 + (size_t)(n0 + r0c) * D + k0c;
  const __bf16* gB11 = pw1 + (size_t)(n0 + r1c) * D + k1c;
  const __bf16* gB30 = pw3 + (size_t)(n0 + r0c) * D + k0c;
  const __bf16* gB31 = pw3 + (size_t)(n0 + r1c) * D + k1c;

  auto issue = [&](int b, int k0) {
    cp16_g2l(&sA [b][dOff0], gA0  + k0);
    cp16_g2l(&sA [b][dOff1], gA1  + k0);
    cp16_g2l(&sB1[b][dOff0], gB10 + k0);
    cp16_g2l(&sB1[b][dOff1], gB11 + k0);
    cp16_g2l(&sB3[b][dOff0], gB30 + k0);
    cp16_g2l(&sB3[b][dOff1], gB31 + k0);
  };

  constexpr int NK = D / BK;   // 24 k-tiles
  issue(0, 0);
  for (int i = 0; i < NK; ++i) {
    const int cur = i & 1;
    wait_async0();       // current tile's async copies done (own counter)
    __syncthreads();     // all waves: data visible, prev buffer free for reuse
    if (i + 1 < NK) issue(cur ^ 1, (i + 1) * BK);
#pragma unroll
    for (int mt = 0; mt < 2; ++mt) {
      v16bf a = ld_frag(&sA[cur][(wm * 32 + mt * 16 + lm) * ST], kb);
#pragma unroll
      for (int nt = 0; nt < 4; ++nt) {
        const int nr = (wn * 64 + nt * 16 + lm) * ST;
        acc1[mt][nt] = wmma_bf16(a, ld_frag(&sB1[cur][nr], kb), acc1[mt][nt]);
        acc3[mt][nt] = wmma_bf16(a, ld_frag(&sB3[cur][nr], kb), acc3[mt][nt]);
      }
    }
  }
  // fused SiLU(x@w1) * (x@w3) epilogue -> bf16 h
#pragma unroll
  for (int mt = 0; mt < 2; ++mt)
#pragma unroll
    for (int nt = 0; nt < 4; ++nt)
#pragma unroll
      for (int v = 0; v < 8; ++v) {
        const int mrow = row0 + wm * 32 + mt * 16 + hi * 8 + v;
        if (mrow < cnt) {
          float a1 = acc1[mt][nt][v];
          float hv = (a1 / (1.f + __expf(-a1))) * acc3[mt][nt][v];
          const int n = n0 + wn * 64 + nt * 16 + lm;
          hb[(size_t)(base + mrow) * H + n] = (__bf16)hv;
        }
      }
}

// ---------------- GEMM2: out[t] += gate * (h @ w2[e]) ----------------
__global__ __launch_bounds__(256) void moe_gemm2(
    const __bf16* __restrict__ hb, const __bf16* __restrict__ w2b,
    const int* __restrict__ btok, const float* __restrict__ bgate,
    const int* __restrict__ counts, const int* __restrict__ offs,
    float* __restrict__ out) {
  __shared__ __attribute__((aligned(16))) __bf16 sA[2][BM * ST];
  __shared__ __attribute__((aligned(16))) __bf16 sB[2][BM * ST];
  __shared__ int   s_tok[BM];
  __shared__ float s_gate[BM];

  const int e    = blockIdx.x / TILES_M;
  const int row0 = (blockIdx.x % TILES_M) * BM;
  const int cnt  = counts[e];
  if (row0 >= cnt) return;
  const int base = offs[e];
  const int n0   = blockIdx.y * BN;   // over D (6 tiles)
  const int tid  = threadIdx.x;

  for (int i = tid; i < BM; i += 256) {
    const bool ok = (row0 + i < cnt);
    s_tok[i]  = ok ? btok [e * NT + row0 + i] : 0;
    s_gate[i] = ok ? bgate[e * NT + row0 + i] : 0.f;
  }
  __syncthreads();

  const int wave = tid >> 5, lane = tid & 31;
  const int wm = wave & 3, wn = wave >> 2;      // 4 x 2 wave grid over 128x128
  const int lm = lane & 15, hi = lane >> 4;
  const int kb = hi * 8;
  v8f acc[2][4] = {};
  const __bf16* pw2 = w2b + (size_t)e * D * H;  // [D][H]

  const int c0 = tid, c1 = tid + 256;
  const int r0c = c0 >> 2, k0c = (c0 & 3) * 8;
  const int r1c = c1 >> 2, k1c = (c1 & 3) * 8;
  const int dOff0 = r0c * ST + k0c, dOff1 = r1c * ST + k1c;
  const int gr0 = imin(base + row0 + r0c, NT * TOPK - 1);
  const int gr1 = imin(base + row0 + r1c, NT * TOPK - 1);
  const __bf16* gA0 = hb + (size_t)gr0 * H + k0c;
  const __bf16* gA1 = hb + (size_t)gr1 * H + k1c;
  const __bf16* gB0 = pw2 + (size_t)(n0 + r0c) * H + k0c;
  const __bf16* gB1 = pw2 + (size_t)(n0 + r1c) * H + k1c;

  auto issue = [&](int b, int k0) {
    cp16_g2l(&sA[b][dOff0], gA0 + k0);
    cp16_g2l(&sA[b][dOff1], gA1 + k0);
    cp16_g2l(&sB[b][dOff0], gB0 + k0);
    cp16_g2l(&sB[b][dOff1], gB1 + k0);
  };

  constexpr int NK = H / BK;   // 64 k-tiles
  issue(0, 0);
  for (int i = 0; i < NK; ++i) {
    const int cur = i & 1;
    wait_async0();
    __syncthreads();
    if (i + 1 < NK) issue(cur ^ 1, (i + 1) * BK);
#pragma unroll
    for (int mt = 0; mt < 2; ++mt) {
      v16bf a = ld_frag(&sA[cur][(wm * 32 + mt * 16 + lm) * ST], kb);
#pragma unroll
      for (int nt = 0; nt < 4; ++nt)
        acc[mt][nt] = wmma_bf16(a, ld_frag(&sB[cur][(wn * 64 + nt * 16 + lm) * ST], kb),
                                acc[mt][nt]);
    }
  }
#pragma unroll
  for (int mt = 0; mt < 2; ++mt)
#pragma unroll
    for (int nt = 0; nt < 4; ++nt)
#pragma unroll
      for (int v = 0; v < 8; ++v) {
        const int ml = wm * 32 + mt * 16 + hi * 8 + v;
        if (row0 + ml < cnt) {
          const int n = n0 + wn * 64 + nt * 16 + lm;
          const float val = acc[mt][nt][v] * s_gate[ml];
          unsafeAtomicAdd(&out[(size_t)s_tok[ml] * D + n], val);
        }
      }
}

// ---------------- host launch ----------------
extern "C" void kernel_launch(void* const* d_in, const int* in_sizes, int n_in,
                              void* d_out, int out_size, void* d_ws, size_t ws_size,
                              hipStream_t stream) {
  const float* x     = (const float*)d_in[0];
  const float* noise = (const float*)d_in[1];
  const float* wg    = (const float*)d_in[2];
  const float* bg    = (const float*)d_in[3];
  const float* wn    = (const float*)d_in[4];
  const float* bn    = (const float*)d_in[5];
  const float* w1    = (const float*)d_in[6];
  const float* w3    = (const float*)d_in[7];
  const float* w2    = (const float*)d_in[8];
  float* out = (float*)d_out;

  // workspace carve-out (256B aligned)
  char* ws = (char*)d_ws;
  size_t off = 0;
  auto carve = [&](size_t bytes) { char* p = ws + off; off = (off + bytes + 255) & ~(size_t)255; return p; };
  int*    counts = (int*)   carve(E * sizeof(int));
  int*    offs   = (int*)   carve(E * sizeof(int));
  int*    btok   = (int*)   carve((size_t)E * NT * sizeof(int));
  float*  bgate  = (float*) carve((size_t)E * NT * sizeof(float));
  __bf16* xb     = (__bf16*)carve((size_t)NT * D * 2);
  __bf16* w1b    = (__bf16*)carve((size_t)E * D * H * 2);   // [e][h][d]
  __bf16* w3b    = (__bf16*)carve((size_t)E * D * H * 2);   // [e][h][d]
  __bf16* w2b    = (__bf16*)carve((size_t)E * D * H * 2);   // [e][d][h]
  __bf16* hb     = (__bf16*)carve((size_t)NT * TOPK * H * 2);
  (void)off; (void)ws_size; (void)in_sizes; (void)n_in; (void)out_size;

  (void)hipMemsetAsync(counts, 0, E * sizeof(int), stream);
  (void)hipMemsetAsync(out, 0, (size_t)NT * D * sizeof(float), stream);

  cvt_x_kernel<<<(NT * D) / (256 * 8), 256, 0, stream>>>(x, xb);
  transpose_cvt_kernel<<<dim3(H / 32, D / 32, E), dim3(32, 8), 0, stream>>>(w1, w1b, D, H);
  transpose_cvt_kernel<<<dim3(H / 32, D / 32, E), dim3(32, 8), 0, stream>>>(w3, w3b, D, H);
  transpose_cvt_kernel<<<dim3(D / 32, H / 32, E), dim3(32, 8), 0, stream>>>(w2, w2b, H, D);

  router_kernel<<<NT / 8, 256, 0, stream>>>(x, noise, wg, bg, wn, bn, counts, btok, bgate);
  offsets_kernel<<<1, 32, 0, stream>>>(counts, offs);

  moe_gemm1<<<dim3(E * TILES_M, H / BN), 256, 0, stream>>>(xb, w1b, w3b, btok, counts, offs, hb);
  moe_gemm2<<<dim3(E * TILES_M, D / BN), 256, 0, stream>>>(hb, w2b, btok, bgate, counts, offs, out);
}
